// MyAttention_56796647522368
// MI455X (gfx1250) — matrix operand order repaired and checked by reading, not compiled, not run on hardware
//
#include <hip/hip_runtime.h>
#include <hip/hip_bf16.h>
#include <math.h>

#define NPATCH 1024
#define CDIM   4096
#define NEGV   (-1e30f)

typedef __attribute__((ext_vector_type(16))) _Float16 v16h;
typedef __attribute__((ext_vector_type(8)))  _Float16 v8h;
typedef __attribute__((ext_vector_type(8)))  float    v8f;

// ---------------------------------------------------------------------------
// Kernel 1: flag = mask (nearest 32x32 -> 32x32 is identity),
//           rn_p[n] = rsqrt(sum_c gen[c,n]^2), rn_k[n] = rsqrt(sum_c kno^2).
// Storing reciprocal sqrts here keeps the GEMM epilogue branch-free
// (1/sqrt(a*b) == rsqrt(a)*rsqrt(b) for positive norms).
// ---------------------------------------------------------------------------
__global__ __launch_bounds__(256)
void norms_flag(const float* __restrict__ gen, const float* __restrict__ kno,
                const float* __restrict__ mask,
                float* __restrict__ rn_p, float* __restrict__ rn_k,
                float* __restrict__ flag)
{
    int n = blockIdx.x * blockDim.x + threadIdx.x;
    if (n >= NPATCH) return;
    float sp = 0.f, sk = 0.f;
    for (int c = 0; c < CDIM; ++c) {
        float g = gen[(size_t)c * NPATCH + n];
        float k = kno[(size_t)c * NPATCH + n];
        sp += g * g;
        sk += k * k;
    }
    rn_p[n] = rsqrtf(sp);
    rn_k[n] = rsqrtf(sk);
    flag[n] = mask[n];
}

// ---------------------------------------------------------------------------
// Kernel 2: transpose-convert fp32 (C,N) -> f16 (N,C) via LDS tile.
// Gives both WMMA operands contiguous-over-K rows (NT GEMM).
// ---------------------------------------------------------------------------
__global__ __launch_bounds__(256)
void cvt_transpose(const float* __restrict__ src, _Float16* __restrict__ dst)
{
    __shared__ float tile[32][33];
    const int cBlock = blockIdx.x * 32;
    const int nBlock = blockIdx.y * 32;
    const int tx = threadIdx.x & 31;
    const int ty = threadIdx.x >> 5;              // 32x8 threads
    #pragma unroll
    for (int i = 0; i < 32; i += 8)
        tile[ty + i][tx] = src[(size_t)(cBlock + ty + i) * NPATCH + nBlock + tx];
    __syncthreads();
    #pragma unroll
    for (int i = 0; i < 32; i += 8)
        dst[(size_t)(nBlock + ty + i) * CDIM + cBlock + tx] =
            (_Float16)tile[tx][ty + i];
}

// ---------------------------------------------------------------------------
// Kernel 3: both cosine matrices in one launch (blockIdx.z selects).
//   z=0: cos0 = norm(P @ Kp^T), keep columns with flag==0
//   z=1: cos1 = norm(P @ P^T),  keep columns with flag==1
// V_WMMA_F32_16X16X32_F16; A and the selected B are (N,C) f16 row-major.
// Block = 4 waves (wave32); block tile 64x128; wave tile 32x64 (2x4 accums):
// 12 b128 loads per 8 WMMAs per k-step.
//
// Per-lane fragment addressing (this kernel's contiguous K order k = 16h + s, same for A and B; not the ISA table order):
//   A 16x32: lane m%16 = row; chunks of 8 halfs at K-offsets h*16 and h*16+8
//            (h = lane/16), packed as vector elements 0..7 / 8..15.
//   B 32x16: lane n%16 = column; 16 contiguous halfs at K-offset h*16.
// ---------------------------------------------------------------------------
__global__ __launch_bounds__(128)
void gemm_cos(const _Float16* __restrict__ Pb, const _Float16* __restrict__ Kb,
              const float* __restrict__ rn_p, const float* __restrict__ rn_k,
              const float* __restrict__ flag,
              float* __restrict__ cos0, float* __restrict__ cos1)
{
    const int z = blockIdx.z;
    const _Float16* __restrict__ A = Pb;
    const _Float16* __restrict__ B = z ? Pb : Kb;
    const float* __restrict__ rnA  = rn_p;
    const float* __restrict__ rnB  = z ? rn_p : rn_k;
    float* __restrict__ out        = z ? cos1 : cos0;
    const float keep               = z ? 1.0f : 0.0f;

    const int lane = threadIdx.x & 31;
    const int w    = threadIdx.x >> 5;            // wave 0..3
    const int rowBase = blockIdx.y * 64 + (w & 1) * 32;
    const int colBase = blockIdx.x * 128 + (w >> 1) * 64;
    const int m16 = lane & 15;
    const int h   = lane >> 4;

    const v8f zero = {0.f, 0.f, 0.f, 0.f, 0.f, 0.f, 0.f, 0.f};
    v8f acc[2][4];
    #pragma unroll
    for (int i = 0; i < 2; ++i)
        #pragma unroll
        for (int j = 0; j < 4; ++j) acc[i][j] = zero;

    for (int k = 0; k < CDIM; k += 32) {
        v16h a[2], b[4];
        #pragma unroll
        for (int i = 0; i < 2; ++i) {
            const _Float16* pa =
                A + (size_t)(rowBase + i * 16 + m16) * CDIM + k + h * 16;
            v8h lo = *(const v8h*)pa;
            v8h hi = *(const v8h*)(pa + 8);
            #pragma unroll
            for (int e = 0; e < 8; ++e) { a[i][e] = lo[e]; a[i][e + 8] = hi[e]; }
        }
        #pragma unroll
        for (int j = 0; j < 4; ++j) {
            const _Float16* pb =
                B + (size_t)(colBase + j * 16 + m16) * CDIM + k + h * 16;
            b[j] = *(const v16h*)pb;              // 32B contiguous
        }
        #pragma unroll
        for (int i = 0; i < 2; ++i)
            #pragma unroll
            for (int j = 0; j < 4; ++j)
                acc[i][j] = __builtin_amdgcn_wmma_f32_16x16x32_f16(
                    false, a[i], false, b[j], (short)0, acc[i][j], false, false);
    }

    // Epilogue: C/D layout — VGPR r, lanes0-15: (M=r, N=lane); lanes16-31: (M=8+r).
    // Branch-free: scale by precomputed rsqrt norms, cndmask the column mask.
    #pragma unroll
    for (int j = 0; j < 4; ++j) {
        const int n    = colBase + j * 16 + m16;
        const bool on  = (flag[n] == keep);
        const float rb = rnB[n];
        #pragma unroll
        for (int i = 0; i < 2; ++i) {
            #pragma unroll
            for (int r = 0; r < 8; ++r) {
                const int m  = rowBase + i * 16 + r + h * 8;
                const float v = acc[i][j][r] * rnA[m] * rb;
                out[(size_t)m * NPATCH + n] = on ? v : NEGV;
            }
        }
    }
}

// ---------------------------------------------------------------------------
// Kernel 4: per-row top-2 (value + global index), one wave32 per row.
// Tie-break toward lower index (matches lax.top_k).
// ---------------------------------------------------------------------------
__device__ __forceinline__ bool better(float va, int ia, float vb, int ib)
{
    return (va > vb) || (va == vb && ia < ib);
}

__global__ __launch_bounds__(32)
void topk2(const float* __restrict__ cos0, const float* __restrict__ cos1,
           float* __restrict__ w0, int* __restrict__ i0,
           float* __restrict__ w1, int* __restrict__ i1)
{
    const float* cosM = (blockIdx.y == 0) ? cos0 : cos1;
    float*       wOut = (blockIdx.y == 0) ? w0 : w1;
    int*         iOut = (blockIdx.y == 0) ? i0 : i1;
    const int row  = blockIdx.x;
    const int lane = threadIdx.x;
    const float* r = cosM + (size_t)row * NPATCH;

    float b1 = -__builtin_huge_valf(); int j1 = 0x7fffffff;
    float b2 = -__builtin_huge_valf(); int j2 = 0x7fffffff;
    for (int c = lane; c < NPATCH; c += 32) {
        float v = r[c];
        if (better(v, c, b1, j1)) { b2 = b1; j2 = j1; b1 = v; j1 = c; }
        else if (better(v, c, b2, j2)) { b2 = v; j2 = c; }
    }
    for (int off = 16; off; off >>= 1) {
        float ob1 = __shfl_xor(b1, off); int oj1 = __shfl_xor(j1, off);
        float ob2 = __shfl_xor(b2, off); int oj2 = __shfl_xor(j2, off);
        if (better(ob1, oj1, b1, j1)) {
            if (better(b1, j1, ob2, oj2)) { b2 = b1; j2 = j1; }
            else                          { b2 = ob2; j2 = oj2; }
            b1 = ob1; j1 = oj1;
        } else if (better(ob1, oj1, b2, j2)) {
            b2 = ob1; j2 = oj1;
        }
    }
    if (lane == 0) {
        wOut[row * 2]     = b1; wOut[row * 2 + 1] = b2;
        iOut[row * 2]     = j1; iOut[row * 2 + 1] = j2;
    }
}

// ---------------------------------------------------------------------------
// Kernel 5: masked-row means of the 4 top-k columns -> softmax -> weights[4].
// ---------------------------------------------------------------------------
__global__ __launch_bounds__(256)
void mk_weights(const float* __restrict__ w0, const float* __restrict__ w1,
                const float* __restrict__ flag, float* __restrict__ weights)
{
    __shared__ float red[5][256];
    const int t = threadIdx.x;
    float s[5] = {0.f, 0.f, 0.f, 0.f, 0.f};
    for (int n = t; n < NPATCH; n += 256) {
        float f = flag[n];
        s[0] += f * w0[n * 2];
        s[1] += f * w0[n * 2 + 1];
        s[2] += f * w1[n * 2];
        s[3] += f * w1[n * 2 + 1];
        s[4] += f;
    }
    for (int q = 0; q < 5; ++q) red[q][t] = s[q];
    __syncthreads();
    for (int off = 128; off; off >>= 1) {
        if (t < off)
            for (int q = 0; q < 5; ++q) red[q][t] += red[q][t + off];
        __syncthreads();
    }
    if (t == 0) {
        float nm = red[4][0];
        float mean[4], mx = -__builtin_huge_valf();
        for (int q = 0; q < 4; ++q) {
            mean[q] = red[q][0] / nm;
            mx = fmaxf(mx, mean[q]);
        }
        float se = 0.f;
        for (int q = 0; q < 4; ++q) { mean[q] = __expf(mean[q] - mx); se += mean[q]; }
        for (int q = 0; q < 4; ++q) weights[q] = mean[q] / se;
    }
}

// ---------------------------------------------------------------------------
// Kernel 6: out = concat([gen, kno, rtn], channel axis).
// rtn[n,c] = sum_q weights[q] * ( flag[n] ? src_q[cols_q[n], c] : 0 )
//            + (n==0) * weights[q] * src_q[0, c] * dup_q  (ind[0,0]=1 artifact)
// stored channel-major like the inputs: out[2CN + c*N + n].
// ---------------------------------------------------------------------------
__global__ __launch_bounds__(256)
void finalize(const float* __restrict__ gen, const float* __restrict__ kno,
              const float* __restrict__ flag,
              const int* __restrict__ i0, const int* __restrict__ i1,
              const float* __restrict__ weights, float* __restrict__ out)
{
    const size_t CN  = (size_t)CDIM * NPATCH;
    size_t idx = (size_t)blockIdx.x * blockDim.x + threadIdx.x;
    if (idx >= CN) return;
    const int    n    = (int)(idx & (NPATCH - 1));
    const size_t crow = idx - (size_t)n;          // c * NPATCH

    out[idx]      = gen[idx];
    out[CN + idx] = kno[idx];

    const float f     = flag[n];
    const float flag0 = flag[0];
    float r = 0.f;
    #pragma unroll
    for (int q = 0; q < 4; ++q) {
        const float* src = (q < 2) ? kno : gen;
        const int*   im  = (q < 2) ? i0  : i1;
        const int col  = im[n * 2 + (q & 1)];
        const int col0 = im[(q & 1)];
        float v = (f == 1.0f) ? src[crow + col] : 0.f;
        if (n == 0) {
            float dup = ((flag0 == 1.0f) && (col0 == 0)) ? 0.f : 1.f;
            v += src[crow] * dup;
        }
        r += weights[q] * v;
    }
    out[2 * CN + idx] = r;
}

// ---------------------------------------------------------------------------
extern "C" void kernel_launch(void* const* d_in, const int* in_sizes, int n_in,
                              void* d_out, int out_size, void* d_ws, size_t ws_size,
                              hipStream_t stream)
{
    const float* gen  = (const float*)d_in[0];   // (1,4096,32,32)
    const float* kno  = (const float*)d_in[1];   // (1,4096,32,32)
    const float* mask = (const float*)d_in[2];   // (1,1,32,32)
    float* out = (float*)d_out;                  // (1,12288,32,32)

    char* ws = (char*)d_ws;
    size_t off = 0;
    auto alloc = [&](size_t bytes) -> void* {
        void* p = ws + off;
        off += (bytes + 255) & ~(size_t)255;
        return p;
    };
    _Float16* Pb   = (_Float16*)alloc((size_t)NPATCH * CDIM * sizeof(_Float16));
    _Float16* Kb   = (_Float16*)alloc((size_t)NPATCH * CDIM * sizeof(_Float16));
    float*    cos0 = (float*)alloc((size_t)NPATCH * NPATCH * sizeof(float));
    float*    cos1 = (float*)alloc((size_t)NPATCH * NPATCH * sizeof(float));
    float*    rn_p = (float*)alloc(NPATCH * sizeof(float));
    float*    rn_k = (float*)alloc(NPATCH * sizeof(float));
    float*    flag = (float*)alloc(NPATCH * sizeof(float));
    float*    w0   = (float*)alloc(NPATCH * 2 * sizeof(float));
    float*    w1   = (float*)alloc(NPATCH * 2 * sizeof(float));
    int*      i0   = (int*)alloc(NPATCH * 2 * sizeof(int));
    int*      i1   = (int*)alloc(NPATCH * 2 * sizeof(int));
    float*    wts  = (float*)alloc(4 * sizeof(float));

    norms_flag<<<NPATCH / 256, 256, 0, stream>>>(gen, kno, mask, rn_p, rn_k, flag);

    dim3 gct(CDIM / 32, NPATCH / 32);
    cvt_transpose<<<gct, 256, 0, stream>>>(gen, Pb);
    cvt_transpose<<<gct, 256, 0, stream>>>(kno, Kb);

    // Both cosine GEMMs in one grid: 8 x 16 x 2 = 256 blocks of 4 waves.
    dim3 gg(NPATCH / 128, NPATCH / 64, 2);
    gemm_cos<<<gg, 128, 0, stream>>>(Pb, Kb, rn_p, rn_k, flag, cos0, cos1);

    topk2<<<dim3(NPATCH, 2), 32, 0, stream>>>(cos0, cos1, w0, i0, w1, i1);
    mk_weights<<<1, 256, 0, stream>>>(w0, w1, flag, wts);

    const size_t CN = (size_t)CDIM * NPATCH;
    finalize<<<(unsigned)((CN + 255) / 256), 256, 0, stream>>>(gen, kno, flag, i0, i1, wts, out);
}
